// AttentionDecoder_71880572665927
// MI455X (gfx1250) — matrix operand order repaired and checked by reading, not compiled, not run on hardware
//
#include <hip/hip_runtime.h>
#include <hip/hip_bf16.h>
#include <math.h>

// ---------------- problem constants ----------------
constexpr int S   = 4096;   // sequence length
constexpr int H   = 2048;   // hidden dim
constexpr int NF  = 512;    // n_features
constexpr int IN_LSTM = H + NF;   // 2560

// ---------------- CDNA5 WMMA types ----------------
typedef __attribute__((ext_vector_type(16))) __bf16 v16bf;
typedef __attribute__((ext_vector_type(8)))  __bf16 v8bf;
typedef __attribute__((ext_vector_type(8)))  float  v8f;

// Branch-free fast transcendentals (v_exp_f32 + v_rcp_f32, no exec-mask
// divergence). Saturation at +-1 / {0,1} falls out of inf/0 propagation.
__device__ __forceinline__ float fast_tanh(float x) {
    float e = __expf(2.0f * x);                       // v_exp_f32
    return 1.0f - 2.0f * __builtin_amdgcn_rcpf(e + 1.0f);  // v_rcp_f32
}
__device__ __forceinline__ float fast_sigmoid(float x) {
    return __builtin_amdgcn_rcpf(1.0f + __expf(-x));
}

// ============================================================
// 1) f32 -> bf16 conversion of Enc[4096x2048] and the encoder
//    half of W_attn (columns H..2H-1), stored K-contiguous.
// ============================================================
__global__ __launch_bounds__(256) void convert_bf16_kernel(
    const float* __restrict__ enc, const float* __restrict__ W_attn,
    __bf16* __restrict__ enc_bf, __bf16* __restrict__ w_bf)
{
    const long stride = (long)gridDim.x * 256;
    long i = (long)blockIdx.x * 256 + threadIdx.x;
    const long n_enc = (long)S * H;       // 8388608
    const long n_w   = (long)H * H;       // 4194304
    for (long idx = i; idx < n_enc; idx += stride)
        enc_bf[idx] = (__bf16)enc[idx];
    for (long idx = i; idx < n_w; idx += stride) {
        long n = idx >> 11;               // row of W_attn
        long k = idx & (H - 1);           // col within encoder half
        w_bf[idx] = (__bf16)W_attn[n * (2 * H) + H + k];
    }
}

// ============================================================
// 2) zero the atomic accumulators (scores[S] then weighted[H]);
//    ws is not re-poisoned between graph replays -> re-init per call
// ============================================================
__global__ void zero_accum_kernel(float* scores, float* weighted) {
    const int i = blockIdx.x * 256 + threadIdx.x;
    if (i < S) scores[i] = 0.0f;
    else       weighted[i - S] = 0.0f;     // grid covers S + H
}

// ============================================================
// 3) h_bias[n] = b_attn[n] + W_attn[n, 0:H] . h_last
//    (the h_rep half of the concat is row-invariant -> bias)
// ============================================================
__global__ __launch_bounds__(256) void hbias_kernel(
    const float* __restrict__ W_attn, const float* __restrict__ b_attn,
    const float* __restrict__ h_last, float* __restrict__ h_bias)
{
    const int lane = threadIdx.x & 31;
    const int wave = threadIdx.x >> 5;
    const int n = blockIdx.x * 8 + wave;          // grid 256 -> 2048 rows
    const float* wrow = W_attn + (size_t)n * (2 * H);
    float acc = 0.0f;
    for (int k = lane; k < H; k += 32)
        acc = fmaf(wrow[k], h_last[k], acc);
    for (int off = 16; off > 0; off >>= 1)
        acc += __shfl_xor(acc, off, 32);
    if (lane == 0) h_bias[n] = acc + b_attn[n];
}

// ============================================================
// 4) WMMA bf16 GEMM + fused tanh / v-dot / row-score reduce.
//    energy = Enc[ S x H ] @ W_enc^T[ H x H ]   (K = H = 2048)
//    scores[m] += sum_n tanh(energy[m,n] + h_bias[n]) * v[n]
//    wave tile 32x64 (2 M-tiles x 4 N-tiles), WG = 8 waves =
//    64x256, grid (S/64, H/256) = (64, 8).
// ============================================================
__global__ __launch_bounds__(256) void attn_scores_kernel(
    const __bf16* __restrict__ enc_bf,   // [S][H] row-major
    const __bf16* __restrict__ w_bf,     // [H][H] row-major (W rows = B cols)
    const float*  __restrict__ h_bias,   // [H]
    const float*  __restrict__ v_attn,   // [H]
    float* __restrict__ scores)          // [S], atomically accumulated
{
    const int lane = threadIdx.x & 31;
    const int wave = threadIdx.x >> 5;
    const int half = lane >> 4;          // 0: lanes 0-15, 1: lanes 16-31
    const int l16  = lane & 15;

    const int m0 = blockIdx.x * 64 + (wave & 1) * 32;
    const int n0 = blockIdx.y * 256 + (wave >> 1) * 64;

    v8f acc[2][4] = {};

    // A fragment (16-bit A 16x32 layout): lane -> row m0+l16,
    // per-lane K = {0..7}+8*half  and  {16..23}+8*half  (two 16B loads)
    const __bf16* aptr0 = enc_bf + (size_t)(m0 + l16) * H + half * 8;
    const __bf16* aptr1 = aptr0 + (size_t)16 * H;          // second M tile
    // B fragment (32x16, K x N): lane -> col n0+l16,
    // per-lane K = {0..7}+8*half  and  {16..23}+8*half  (two 16B loads)
    const __bf16* bptr  = w_bf + (size_t)(n0 + l16) * H + half * 8;

    for (int k0 = 0; k0 < H; k0 += 32) {
        v16bf a[2], b[4];
        union { v16bf v; v8bf h[2]; } u;
        u.h[0] = *(const v8bf*)(aptr0 + k0);
        u.h[1] = *(const v8bf*)(aptr0 + k0 + 16);
        a[0] = u.v;
        u.h[0] = *(const v8bf*)(aptr1 + k0);
        u.h[1] = *(const v8bf*)(aptr1 + k0 + 16);
        a[1] = u.v;
        #pragma unroll
        for (int j = 0; j < 4; ++j) {
            const __bf16* bj = bptr + (size_t)j * 16 * H + k0;
            u.h[0] = *(const v8bf*)(bj);
            u.h[1] = *(const v8bf*)(bj + 16);
            b[j] = u.v;
        }
        #pragma unroll
        for (int i = 0; i < 2; ++i)
            #pragma unroll
            for (int j = 0; j < 4; ++j)
                acc[i][j] = __builtin_amdgcn_wmma_f32_16x16x32_bf16(
                    false, a[i], false, b[j], (short)0, acc[i][j],
                    false, false);
    }

    // Fused epilogue. C layout: lane half h, VGPR r -> row m0+i*16+8h+r,
    // col n0+j*16+l16. Reduce over the 16 lanes of each half (offsets
    // 1..8 stay inside the half, so rows r and r+8 never mix).
    #pragma unroll
    for (int i = 0; i < 2; ++i) {
        float rowsum[8] = {0.f,0.f,0.f,0.f,0.f,0.f,0.f,0.f};
        #pragma unroll
        for (int j = 0; j < 4; ++j) {
            const int n = n0 + j * 16 + l16;
            const float hb = h_bias[n];
            const float vv = v_attn[n];
            #pragma unroll
            for (int r = 0; r < 8; ++r) {
                float e = fast_tanh(acc[i][j][r] + hb);
                rowsum[r] = fmaf(e, vv, rowsum[r]);
            }
        }
        #pragma unroll
        for (int r = 0; r < 8; ++r) {
            float ssum = rowsum[r];
            for (int off = 1; off < 16; off <<= 1)
                ssum += __shfl_xor(ssum, off, 32);
            if (l16 == 0)
                atomicAdd(&scores[m0 + i * 16 + half * 8 + r], ssum);
        }
    }
}

// ============================================================
// 5) softmax over 4096 scores, single workgroup, LDS reductions
// ============================================================
__global__ __launch_bounds__(1024) void softmax_kernel(
    const float* __restrict__ scores, float* __restrict__ a)
{
    __shared__ float red[1024];
    const int t = threadIdx.x;
    float m = -3.4e38f;
    for (int s = t; s < S; s += 1024) m = fmaxf(m, scores[s]);
    red[t] = m; __syncthreads();
    for (int off = 512; off > 0; off >>= 1) {
        if (t < off) red[t] = fmaxf(red[t], red[t + off]);
        __syncthreads();
    }
    const float mx = red[0];
    __syncthreads();
    float sum = 0.0f;
    for (int s = t; s < S; s += 1024) {
        float e = __expf(scores[s] - mx);   // arg <= 0: no overflow
        a[s] = e;
        sum += e;
    }
    red[t] = sum; __syncthreads();
    for (int off = 512; off > 0; off >>= 1) {
        if (t < off) red[t] += red[t + off];
        __syncthreads();
    }
    const float inv = __builtin_amdgcn_rcpf(red[0]);
    for (int s = t; s < S; s += 1024) a[s] *= inv;
}

// ============================================================
// 6) weighted[h] += sum_{s in block} a[s] * enc[s,h]
//    grid (H/256, S/256) = 128 WGs -> enough waves in flight to
//    pull 33.5 MB at HBM rate; 16 atomicAdds per column total.
// ============================================================
__global__ __launch_bounds__(256) void weighted_kernel(
    const float* __restrict__ a, const float* __restrict__ enc,
    float* __restrict__ weighted)
{
    const int col = blockIdx.x * 256 + threadIdx.x;
    const int s0  = blockIdx.y * 256;
    float acc = 0.0f;
    for (int s = s0; s < s0 + 256; ++s)
        acc = fmaf(a[s], enc[(size_t)s * H + col], acc);
    atomicAdd(&weighted[col], acc);
}

// ============================================================
// 7) LSTM step. WG = 8 waves handles 2 units; wave (u,g) computes
//    gate row g*H+u as a 4608-long dot (x | weighted | h0), then
//    2 threads combine i,f,g,o from LDS into c_n / h_n.
//    Prefetch hints ahead of the 151 MB weight stream
//    (global_prefetch_b8 path).
// ============================================================
__global__ __launch_bounds__(256) void lstm_kernel(
    const float* __restrict__ x, const float* __restrict__ weighted,
    const float* __restrict__ h0, const float* __restrict__ c0,
    const float* __restrict__ W_ih, const float* __restrict__ W_hh,
    const float* __restrict__ b_ih, const float* __restrict__ b_hh,
    float* __restrict__ h_out, float* __restrict__ c_out)
{
    __shared__ float gate_lds[8];
    const int lane = threadIdx.x & 31;
    const int wave = threadIdx.x >> 5;
    const int unit = blockIdx.x * 2 + (wave >> 2);   // grid 1024 -> 2048 units
    const int g    = wave & 3;                       // i, f, g, o
    const size_t r = (size_t)g * H + unit;
    const float* wih = W_ih + r * IN_LSTM;
    const float* whh = W_hh + r * H;
    float acc = 0.0f;
    for (int j = lane; j < NF; j += 32) acc = fmaf(wih[j], x[j], acc);
    for (int j = lane; j < H;  j += 32) {
        __builtin_prefetch(&wih[NF + j + 1024], 0, 0);
        acc = fmaf(wih[NF + j], weighted[j], acc);
    }
    for (int j = lane; j < H;  j += 32) {
        __builtin_prefetch(&whh[j + 1024], 0, 0);
        acc = fmaf(whh[j], h0[j], acc);
    }
    for (int off = 16; off > 0; off >>= 1) acc += __shfl_xor(acc, off, 32);
    if (lane == 0) gate_lds[wave] = acc + b_ih[r] + b_hh[r];
    __syncthreads();
    if (threadIdx.x < 2) {
        const int u = blockIdx.x * 2 + threadIdx.x;
        const float gi = gate_lds[threadIdx.x * 4 + 0];
        const float gf = gate_lds[threadIdx.x * 4 + 1];
        const float gg = gate_lds[threadIdx.x * 4 + 2];
        const float go = gate_lds[threadIdx.x * 4 + 3];
        const float c = fast_sigmoid(gf) * c0[u] + fast_sigmoid(gi) * fast_tanh(gg);
        const float h = fast_sigmoid(go) * fast_tanh(c);
        c_out[u] = c;
        h_out[u] = h;
    }
}

// ============================================================
// 8) out[o] = b_out[o] + W_out[o,:] . cat(h_n, weighted)
// ============================================================
__global__ __launch_bounds__(256) void out_kernel(
    const float* __restrict__ W_out, const float* __restrict__ b_out,
    const float* __restrict__ h_n, const float* __restrict__ weighted,
    float* __restrict__ out)
{
    const int lane = threadIdx.x & 31;
    const int wave = threadIdx.x >> 5;
    const int o = blockIdx.x * 8 + wave;              // grid 64 -> 512 rows
    const float* wrow = W_out + (size_t)o * (2 * H);
    float acc = 0.0f;
    for (int j = lane; j < H; j += 32) acc = fmaf(wrow[j], h_n[j], acc);
    for (int j = lane; j < H; j += 32) acc = fmaf(wrow[H + j], weighted[j], acc);
    for (int off = 16; off > 0; off >>= 1) acc += __shfl_xor(acc, off, 32);
    if (lane == 0) out[o] = acc + b_out[o];
}

// ============================================================
extern "C" void kernel_launch(void* const* d_in, const int* in_sizes, int n_in,
                              void* d_out, int out_size, void* d_ws, size_t ws_size,
                              hipStream_t stream) {
    const float* x        = (const float*)d_in[0];
    const float* in_h     = (const float*)d_in[1];   // [1,1,H] -> h_last == h0
    const float* in_c     = (const float*)d_in[2];
    const float* enc      = (const float*)d_in[3];   // [1,S,H]
    const float* W_attn   = (const float*)d_in[4];   // [H, 2H]
    const float* b_attn   = (const float*)d_in[5];
    const float* v_attn   = (const float*)d_in[6];
    const float* W_ih     = (const float*)d_in[7];   // [4H, H+NF]
    const float* W_hh     = (const float*)d_in[8];   // [4H, H]
    const float* b_ih     = (const float*)d_in[9];
    const float* b_hh     = (const float*)d_in[10];
    const float* W_out    = (const float*)d_in[11];  // [NF, 2H]
    const float* b_out    = (const float*)d_in[12];

    // ---- workspace layout ----
    char* ws = (char*)d_ws;
    __bf16* enc_bf = (__bf16*)ws;                               // S*H*2   = 16 MiB
    __bf16* w_bf   = (__bf16*)(ws + (size_t)S * H * 2);         // H*H*2   =  8 MiB
    float*  h_bias = (float*)(ws + (size_t)S * H * 2 + (size_t)H * H * 2);
    float*  scores   = h_bias + H;        // [S]
    float*  attw     = scores + S;        // [S]
    float*  weighted = attw + S;          // [H]

    // ---- output layout: [ out(NF) | h_n(H) | c_n(H) ] ----
    float* out_p = (float*)d_out;
    float* h_n_p = out_p + NF;
    float* c_n_p = h_n_p + H;

    convert_bf16_kernel<<<1024, 256, 0, stream>>>(enc, W_attn, enc_bf, w_bf);
    zero_accum_kernel<<<(S + H) / 256, 256, 0, stream>>>(scores, weighted);
    hbias_kernel<<<H / 8, 256, 0, stream>>>(W_attn, b_attn, in_h, h_bias);
    attn_scores_kernel<<<dim3(S / 64, H / 256), 256, 0, stream>>>(
        enc_bf, w_bf, h_bias, v_attn, scores);
    softmax_kernel<<<1, 1024, 0, stream>>>(scores, attw);
    weighted_kernel<<<dim3(H / 256, S / 256), 256, 0, stream>>>(attw, enc, weighted);
    lstm_kernel<<<H / 2, 256, 0, stream>>>(x, weighted, in_h, in_c,
                                           W_ih, W_hh, b_ih, b_hh, h_n_p, c_n_p);
    out_kernel<<<NF / 8, 256, 0, stream>>>(W_out, b_out, h_n_p, weighted, out_p);
}
